// STGCNBlock_48541720379458
// MI455X (gfx1250) — matrix operand order repaired and checked by reading, not compiled, not run on hardware
//
#include <hip/hip_runtime.h>

typedef __attribute__((ext_vector_type(16))) _Float16 v16h;
typedef __attribute__((ext_vector_type(8)))  _Float16 v8h;
typedef __attribute__((ext_vector_type(4)))  _Float16 v4h;
typedef __attribute__((ext_vector_type(8)))  float    v8f;

// Problem sizes (compile-time, from reference)
constexpr int B = 4, F = 32, C = 4096, T = 32, E = 65536, KW = 9;
constexpr long CT   = (long)C * T;        // 131072 (c,t) slots per (b)
constexpr long NROW = (long)B * C * T;    // 524288 rows of node-feature matrix
constexpr long N1   = NROW * F;           // 16777216 elements in [b,c,t,f] tensors

// ---------------------------------------------------------------- degree / norm
__global__ void k_deg_init(float* __restrict__ deg) {
    int i = blockIdx.x * 256 + threadIdx.x;
    if (i < C) deg[i] = 1.0f;                      // self-loop fill value 1
}

__global__ void k_deg_scatter(const int* __restrict__ ei, const float* __restrict__ ew,
                              float* __restrict__ deg) {
    int e = blockIdx.x * 256 + threadIdx.x;
    if (e < E) atomicAdd(&deg[ei[E + e]], ew[e]);  // dst row of edge_index
}

__global__ void k_dis(const float* __restrict__ deg, float* __restrict__ dis) {
    int i = blockIdx.x * 256 + threadIdx.x;
    if (i < C) dis[i] = rsqrtf(deg[i]);
}

// ---------------------------------------------------------------- GCN GEMM (WMMA)
// xlin[b,c,t,g] = sum_f x[b,f,c,t] * W[f,g];  f16 inputs, f32 accumulate
__global__ __launch_bounds__(256) void k_gcn_gemm(const float* __restrict__ x,
                                                  const float* __restrict__ W,
                                                  float* __restrict__ xlin) {
    __shared__ __align__(16) _Float16 wl[F][F];    // [g][f] (transposed for B reads)
    int tid = threadIdx.x;
    for (int i = tid; i < F * F; i += 256) {
        int f = i / F, g = i % F;
        wl[g][f] = (_Float16)W[f * F + g];
    }
    __syncthreads();

    int wave = tid >> 5, lane = tid & 31;
    long tile  = (long)blockIdx.x * 8 + wave;      // 16 rows per wave-tile
    long rbase = tile * 16;
    int  m  = lane & 15;
    int  hb = lane >> 4;                           // lane half selects K-block
    long r  = rbase + m;
    int  b  = (int)(r / CT);
    int  rem = (int)(r % CT);
    int  c  = rem / T, t = rem % T;
    const float* xp = x + (long)b * F * CT + (long)c * T + t;   // + f*CT

    v16h a;
    #pragma unroll
    for (int v = 0; v < 8; ++v) {
        int k0 = hb * 16 + 2 * v;
        a[2 * v]     = (_Float16)xp[(long)k0 * CT];
        a[2 * v + 1] = (_Float16)xp[(long)(k0 + 1) * CT];
    }

    float* outp = xlin + rbase * F;
    int n  = lane & 15;
    int kb = hb * 16;
    #pragma unroll
    for (int nt = 0; nt < 2; ++nt) {
        const _Float16* wrow = &wl[n + nt * 16][kb];           // 16 contiguous halves
        v8h blo = *(const v8h*)wrow;                           // ds_load_b128
        v8h bhi = *(const v8h*)(wrow + 8);                     // ds_load_b128
        v16h bm = __builtin_shufflevector(blo, bhi, 0,1,2,3,4,5,6,7,
                                          8,9,10,11,12,13,14,15);
        v8f acc = {};
        acc = __builtin_amdgcn_wmma_f32_16x16x32_f16(false, a, false, bm,
                                                     (short)0, acc, false, false);
        #pragma unroll
        for (int v = 0; v < 8; ++v) {
            int mm = v + hb * 8;
            outp[(long)mm * F + n + nt * 16] = acc[v];
        }
    }
}

// ---------------------------------------------------------------- graph aggregate
__global__ void k_agg_init(const float* __restrict__ xlin, const float* __restrict__ dis,
                           float* __restrict__ agg) {
    long i = (long)blockIdx.x * 256 + threadIdx.x;   // float4 index
    long idx = i * 4;
    if (idx >= N1) return;
    long row = idx / F;                              // (b*C + c)*T + t
    int  c   = (int)((row / T) % C);
    float s  = dis[c]; s *= s;                       // self-loop dis^2
    float4 v = ((const float4*)xlin)[i];
    v.x *= s; v.y *= s; v.z *= s; v.w *= s;
    ((float4*)agg)[i] = v;
}

__global__ void k_edge_scatter(const int* __restrict__ ei, const float* __restrict__ ew,
                               const float* __restrict__ dis, const float* __restrict__ xlin,
                               float* __restrict__ agg) {
    long gid = (long)blockIdx.x * 256 + threadIdx.x;     // over E * B * T
    if (gid >= (long)E * B * T) return;
    int e  = (int)(gid / (B * T));
    int bt = (int)(gid % (B * T));
    int b  = bt / T, t = bt % T;
    int s  = ei[e], d = ei[E + e];
    float nrm = dis[s] * ew[e] * dis[d];
    const float4* s4 = (const float4*)(xlin + (((long)b * C + s) * T + t) * F);
    float*        dp = agg + (((long)b * C + d) * T + t) * F;
    #pragma unroll
    for (int q = 0; q < 8; ++q) {                        // 8 x b128 gather
        float4 v = s4[q];
        atomicAdd(&dp[4 * q + 0], v.x * nrm);
        atomicAdd(&dp[4 * q + 1], v.y * nrm);
        atomicAdd(&dp[4 * q + 2], v.z * nrm);
        atomicAdd(&dp[4 * q + 3], v.w * nrm);
    }
}

// ---------------------------------------------------------------- instance norm stats
// One block per (b,g) group; reduce 131072 elements -> mu, 1/sigma. No atomics.
__global__ __launch_bounds__(256) void k_in_stats(const float* __restrict__ ybuf,
                                                  float* __restrict__ mu,
                                                  float* __restrict__ rs) {
    int grp = blockIdx.x;                    // b*F + g
    int b = grp / F, g = grp % F;
    const float* p = ybuf + (long)b * CT * F + g;
    float s = 0.f, s2 = 0.f;
    for (long i = threadIdx.x; i < CT; i += 256) {
        float v = p[i * F];
        s += v; s2 += v * v;
    }
    __shared__ float sh[256], sh2[256];
    sh[threadIdx.x] = s; sh2[threadIdx.x] = s2;
    __syncthreads();
    for (int o = 128; o > 0; o >>= 1) {
        if (threadIdx.x < o) { sh[threadIdx.x] += sh[threadIdx.x + o];
                               sh2[threadIdx.x] += sh2[threadIdx.x + o]; }
        __syncthreads();
    }
    if (threadIdx.x == 0) {
        float m   = sh[0] / (float)CT;
        float var = sh2[0] / (float)CT - m * m;
        mu[grp] = m;
        rs[grp] = rsqrtf(var + 1e-5f);
    }
}

// normalize + relu, emit f16 (feeds the conv WMMA).  Bias cancels under IN.
__global__ void k_norm_relu_f16(const float* __restrict__ agg, const float* __restrict__ mu,
                                const float* __restrict__ rs, _Float16* __restrict__ y1) {
    long i4 = (long)blockIdx.x * 256 + threadIdx.x;      // float4 index
    long idx = i4 * 4;
    if (idx >= N1) return;
    int g0 = (int)(idx % F);                             // multiple of 4
    int b  = (int)((idx / F) / CT);
    int grp = b * F + g0;
    float4 v = ((const float4*)agg)[i4];
    v4h h;
    h[0] = (_Float16)fmaxf((v.x - mu[grp + 0]) * rs[grp + 0], 0.f);
    h[1] = (_Float16)fmaxf((v.y - mu[grp + 1]) * rs[grp + 1], 0.f);
    h[2] = (_Float16)fmaxf((v.z - mu[grp + 2]) * rs[grp + 2], 0.f);
    h[3] = (_Float16)fmaxf((v.w - mu[grp + 3]) * rs[grp + 3], 0.f);
    *(v4h*)(y1 + idx) = h;                               // 8-byte store
}

// ---------------------------------------------------------------- temporal conv (WMMA)
// y2[b,c,t,co] = sum_{k,ci} y1[b,c,t+k-4,ci] * conv_w[co,ci,0,k]   (zero padded)
// Block = 8 waves = 4 (b,c) slabs x 2 t-tiles. The 4 slabs (8 KB of f16) are
// staged into LDS with gfx1250 async-to-LDS copies, then all 9 taps hit LDS.
__global__ __launch_bounds__(256) void k_tconv(const _Float16* __restrict__ y1,
                                               const float* __restrict__ cw,
                                               float* __restrict__ y2) {
    __shared__ __align__(16) _Float16 wl[KW][F][F];      // [k][co][ci], 18 KB
    __shared__ __align__(16) _Float16 ytile[4][T][F];    // 4 slabs, 8 KB
    int tid = threadIdx.x;

    // async-stage the 4 y1 slabs: 512 chunks of 16 B
    {
        unsigned ldsbase = (unsigned)(uintptr_t)(&ytile[0][0][0]);
        long gsbase = (long)blockIdx.x * 4;              // first slab index
        #pragma unroll
        for (int ch = 0; ch < 2; ++ch) {
            int chunk = tid + ch * 256;                  // 0..511
            int slab  = chunk >> 7;
            int off   = (chunk & 127) * 16;              // byte offset in slab
            unsigned lds = ldsbase + slab * 2048 + off;
            unsigned long long ga =
                (unsigned long long)((const char*)y1 + (gsbase + slab) * 2048 + off);
            asm volatile("global_load_async_to_lds_b128 %0, %1, off"
                         :: "v"(lds), "v"(ga) : "memory");
        }
    }

    // stage conv weights as f16 [k][co][ci] (sync path, 36 KB source)
    for (int i = tid; i < KW * F * F; i += 256) {
        int k = i / (F * F); int rem = i % (F * F);
        int co = rem / F, ci = rem % F;
        wl[k][co][ci] = (_Float16)cw[((long)co * F + ci) * KW + k];
    }

    asm volatile("s_wait_asynccnt 0" ::: "memory");      // async LDS writes visible
    __syncthreads();

    int wave = tid >> 5, lane = tid & 31;
    int slab  = wave >> 1;                               // 0..3
    int ttile = wave & 1;                                // 0..1
    long gs = (long)blockIdx.x * 4 + slab;               // global (b,c) slab index
    int  m  = lane & 15, hb = lane >> 4;
    int  t  = ttile * 16 + m;                            // row within slab

    v8f acc0 = {}, acc1 = {};
    int n  = lane & 15;
    int kb = hb * 16;
    #pragma unroll
    for (int k = 0; k < KW; ++k) {
        int tt = t + k - 4;
        v16h a = {};
        if (tt >= 0 && tt < T) {
            const _Float16* rp = &ytile[slab][tt][0];
            v8h lo = *(const v8h*)(rp + hb * 16);        // ds_load_b128
            v8h hi = *(const v8h*)(rp + hb * 16 + 8);    // ds_load_b128
            a = __builtin_shufflevector(lo, hi, 0,1,2,3,4,5,6,7,
                                        8,9,10,11,12,13,14,15);
        }
        const _Float16* w0 = &wl[k][n][kb];
        const _Float16* w1 = &wl[k][n + 16][kb];
        v16h b0 = __builtin_shufflevector(*(const v8h*)w0, *(const v8h*)(w0 + 8),
                                          0,1,2,3,4,5,6,7,8,9,10,11,12,13,14,15);
        v16h b1 = __builtin_shufflevector(*(const v8h*)w1, *(const v8h*)(w1 + 8),
                                          0,1,2,3,4,5,6,7,8,9,10,11,12,13,14,15);
        acc0 = __builtin_amdgcn_wmma_f32_16x16x32_f16(false, a, false, b0,
                                                      (short)0, acc0, false, false);
        acc1 = __builtin_amdgcn_wmma_f32_16x16x32_f16(false, a, false, b1,
                                                      (short)0, acc1, false, false);
    }
    float* outp = y2 + (gs * T + ttile * 16) * F;
    #pragma unroll
    for (int v = 0; v < 8; ++v) {
        int mm = v + hb * 8;
        outp[(long)mm * F + n]      = acc0[v];
        outp[(long)mm * F + n + 16] = acc1[v];
    }
}

// ---------------------------------------------------------------- final fuse
// out[b,f,c,t] = relu( relu(IN(y2))[b,c,t,f] + x[b,f,c,t] )
__global__ void k_final(const float* __restrict__ y2, const float* __restrict__ mu,
                        const float* __restrict__ rs, const float* __restrict__ x,
                        float* __restrict__ out) {
    long i = (long)blockIdx.x * 256 + threadIdx.x;   // flat [b,f,c,t]
    if (i >= N1) return;
    int t = (int)(i % T); long q = i / T;
    int c = (int)(q % C); q /= C;
    int f = (int)(q % F);
    int b = (int)(q / F);
    float v = y2[(((long)b * C + c) * T + t) * F + f];
    int grp = b * F + f;
    v = fmaxf((v - mu[grp]) * rs[grp], 0.f);
    out[i] = fmaxf(v + x[i], 0.f);
}

// ---------------------------------------------------------------- launch
extern "C" void kernel_launch(void* const* d_in, const int* in_sizes, int n_in,
                              void* d_out, int out_size, void* d_ws, size_t ws_size,
                              hipStream_t stream) {
    const float* x   = (const float*)d_in[0];
    const int*   ei  = (const int*)d_in[1];     // [2,E]
    const float* ew  = (const float*)d_in[2];
    const float* W   = (const float*)d_in[3];
    // d_in[4] = gcn_b, d_in[6] = conv_b: cancel under affine-free InstanceNorm
    const float* cw  = (const float*)d_in[5];
    float*       out = (float*)d_out;

    char* ws = (char*)d_ws;
    float*    xlin = (float*)ws;                          // N1 floats (67 MB)
    float*    agg  = (float*)(ws + N1 * sizeof(float));   // N1 floats (67 MB)
    _Float16* y1   = (_Float16*)xlin;                     // reuse: f16, 33.5 MB
    float*    y2   = agg;                                 // reuse after IN pass 1
    char* tail = ws + 2 * N1 * sizeof(float);
    float* deg = (float*)tail;                            // C
    float* dis = deg + C;                                 // C
    float* mu1 = dis + C;  float* rs1 = mu1 + B * F;
    float* mu2 = rs1 + B * F; float* rs2 = mu2 + B * F;

    // 1) gcn_norm
    k_deg_init   <<<(C + 255) / 256, 256, 0, stream>>>(deg);
    k_deg_scatter<<<E / 256,        256, 0, stream>>>(ei, ew, deg);
    k_dis        <<<(C + 255) / 256, 256, 0, stream>>>(deg, dis);

    // 2) x_lin = x @ W  (WMMA)
    k_gcn_gemm<<<(int)(NROW / 16 / 8), 256, 0, stream>>>(x, W, xlin);

    // 3) agg = self-loop + edge scatter
    k_agg_init    <<<(int)(N1 / 4 / 256), 256, 0, stream>>>(xlin, dis, agg);
    k_edge_scatter<<<(int)((long)E * B * T / 256), 256, 0, stream>>>(ei, ew, dis, xlin, agg);

    // 4) IN -> ReLU -> f16
    k_in_stats     <<<B * F, 256, 0, stream>>>(agg, mu1, rs1);
    k_norm_relu_f16<<<(int)(N1 / 4 / 256), 256, 0, stream>>>(agg, mu1, rs1, y1);

    // 5) temporal conv (WMMA, async LDS staging)
    k_tconv<<<(int)(B * (long)C / 4), 256, 0, stream>>>(y1, cw, y2);

    // 6) IN -> ReLU -> +residual -> ReLU (transposed write to NCHW)
    k_in_stats<<<B * F, 256, 0, stream>>>(y2, mu2, rs2);
    k_final   <<<(int)(N1 / 256), 256, 0, stream>>>(y2, mu2, rs2, x, out);
}
